// TransformerEncoder_89352499626039
// MI455X (gfx1250) — matrix operand order repaired and checked by reading, not compiled, not run on hardware
//
#include <hip/hip_runtime.h>
#include <hip/hip_bf16.h>
#include <math.h>
#include <stdint.h>

typedef __attribute__((ext_vector_type(16))) _Float16 v16h;
typedef __attribute__((ext_vector_type(8)))  float    v8f;

constexpr int Bc = 2, Sc = 2048, Dc = 256, Hc = 8, FFc = 1024, Lc = 4;
constexpr int DKc = 32;
constexpr int BS  = Bc * Sc;          // 4096 token rows
constexpr float EPSc = 1e-5f;

union F16x16 { v16h v; unsigned u[8]; };

// ---------------------------------------------------------------------------
// h = x + positional encoding
// ---------------------------------------------------------------------------
__global__ void k_add_pe(const float* __restrict__ x, float* __restrict__ h) {
  int idx = blockIdx.x * blockDim.x + threadIdx.x;
  if (idx >= BS * Dc) return;
  int d   = idx % Dc;
  int pos = (idx / Dc) % Sc;
  int d2  = d & ~1;
  float div = __expf(-9.210340371976184f * (float)d2 / (float)Dc); // exp(-ln(1e4)*d2/D)
  float ang = (float)pos * div;
  float pe  = (d & 1) ? __cosf(ang) : __sinf(ang);
  h[idx] = x[idx] + pe;
}

// ---------------------------------------------------------------------------
// One-time per-launch weight prep: W[K,N] fp32 -> Wt[N,K] f16 (transposed).
// Makes GEMM B-tiles async-DMA-able and B-fragment reads contiguous.
// ---------------------------------------------------------------------------
__global__ void k_wt_f16(const float* __restrict__ W, _Float16* __restrict__ Wt,
                         int K, int N) {
  int idx = blockIdx.x * blockDim.x + threadIdx.x;
  if (idx >= K * N) return;
  int n = idx / K;
  int k = idx % K;
  Wt[(size_t)n * K + k] = (_Float16)W[(size_t)k * N + n];
}

// ---------------------------------------------------------------------------
// C[M,N] = act(A[M,K] @ W[K,N] + bias[N]) with Wt = W^T in f16 [N,K].
// 64x64 tile, 4 waves. B tile staged via GLOBAL_LOAD_ASYNC_TO_LDS_B128
// (ASYNCcnt), A tile converted f32->f16 by VALU. WMMA f16 -> f32 acc.
// All dims are multiples of 64/32 here -> no bounds checks.
// ---------------------------------------------------------------------------
__global__ __launch_bounds__(128) void k_gemm(const float* __restrict__ A,
                                              const _Float16* __restrict__ Wt,
                                              const float* __restrict__ bias,
                                              float* __restrict__ C,
                                              int M, int N, int K, int relu) {
  __shared__ _Float16 As[64 * 40];   // [row][k]  stride 40
  __shared__ _Float16 Bs[64 * 40];   // [n][k]    stride 40 (already transposed)

  const int tid   = threadIdx.x;
  const int lane  = tid & 31;
  const int wave  = tid >> 5;
  const int tileM = blockIdx.y * 64;
  const int tileN = blockIdx.x * 64;
  const int waveM = (wave >> 1) * 32;
  const int waveN = (wave & 1) * 32;
  const int mrow  = lane & 15;
  const int kgrp  = lane >> 4;
  (void)M;

  v8f acc[2][2] = {};

  for (int k0 = 0; k0 < K; k0 += 32) {
    // ---- async DMA the Wt tile: 64 rows x 64B, 16B chunks per lane ----
    for (int t = tid; t < 256; t += 128) {
      int r = t >> 2, c = t & 3;
      unsigned ldsa = (unsigned)(uintptr_t)(&Bs[r * 40 + c * 8]);
      uint64_t ga   = (uint64_t)(uintptr_t)(Wt + (size_t)(tileN + r) * K + k0 + c * 8);
      asm volatile("global_load_async_to_lds_b128 %0, %1, off"
                   :: "v"(ldsa), "v"(ga) : "memory");
    }
    // ---- stage A tile 64x32 (fp32 -> f16) ----
    for (int t = tid; t < 64 * 32; t += 128) {
      int r = t >> 5, c = t & 31;
      As[r * 40 + c] = (_Float16)A[(size_t)(tileM + r) * K + k0 + c];
    }
    // prefetch next A K-slab
    if (k0 + 32 < K)
      __builtin_prefetch(&A[(size_t)(tileM + (tid & 63)) * K + k0 + 32], 0, 1);

    asm volatile("s_wait_asynccnt 0x0" ::: "memory");
    __syncthreads();

    const unsigned* pa = (const unsigned*)As;
    const unsigned* pb = (const unsigned*)Bs;

    F16x16 af[2], bf[2];
    for (int mi = 0; mi < 2; ++mi) {
      int row = waveM + mi * 16 + mrow;
      for (int j = 0; j < 8; ++j) {
        // A-layout: K = kgrp*16 + 2j (+sub via pair load)
        int kk = (kgrp << 4) + (j << 1);
        af[mi].u[j] = pa[row * 20 + (kk >> 1)];
      }
    }
    for (int ni = 0; ni < 2; ++ni) {
      int coln = waveN + ni * 16 + mrow;
      for (int j = 0; j < 8; ++j) {
        // B-layout: K = kgrp*16 + 2j (+sub) , N = lane&15
        bf[ni].u[j] = pb[coln * 20 + (kgrp << 3) + j];
      }
    }
    for (int mi = 0; mi < 2; ++mi)
      for (int ni = 0; ni < 2; ++ni)
        acc[mi][ni] = __builtin_amdgcn_wmma_f32_16x16x32_f16(
            false, af[mi].v, false, bf[ni].v, (short)0, acc[mi][ni], false, false);
    __syncthreads();
  }

  // epilogue: C layout row = kgrp*8 + r, col = lane&15
  for (int mi = 0; mi < 2; ++mi)
    for (int ni = 0; ni < 2; ++ni)
      for (int r = 0; r < 8; ++r) {
        int row = tileM + waveM + mi * 16 + (kgrp << 3) + r;
        int col = tileN + waveN + ni * 16 + mrow;
        float val = acc[mi][ni][r] + bias[col];
        if (relu) val = fmaxf(val, 0.0f);
        C[(size_t)row * N + col] = val;
      }
}

// ---------------------------------------------------------------------------
// Flash attention: 4 waves / block, 64 query rows of one (b,h) per block.
// K and V tiles (32 keys) staged once per block in LDS f16:
//   sK [key][dk]  -> K^T fragment reads contiguous
//   sV [dk][key]  -> V fragment reads contiguous
// q/k/v are [BS, D] with head h_ in columns [h_*32, h_*32+32).
// ---------------------------------------------------------------------------
__global__ __launch_bounds__(128) void k_attn(const float* __restrict__ q,
                                              const float* __restrict__ k,
                                              const float* __restrict__ v,
                                              float* __restrict__ ctx) {
  __shared__ _Float16 sK[32 * 40];   // [key][dk]
  __shared__ _Float16 sV[32 * 40];   // [dk][key]
  __shared__ _Float16 sP[64 * 36];   // per-wave 16x32 P tiles

  const int tid  = threadIdx.x;
  const int lane = tid & 31;
  const int wave = tid >> 5;
  const int nq   = Sc / 64;              // 32 q-blocks per (b,h)
  const int qt   = blockIdx.x % nq;
  const int bh   = blockIdx.x / nq;
  const int b_   = bh / Hc;
  const int h_   = bh % Hc;
  const int hoff = h_ * DKc;
  const int col  = lane & 15;
  const int g    = lane >> 4;
  const int base0 = b_ * Sc;
  const int qrow0 = qt * 64 + wave * 16;

  // Q fragment (A-layout 16x32), scale 1/sqrt(32) folded in
  F16x16 aq;
  {
    const float* qr = q + (size_t)(base0 + qrow0 + col) * Dc + hoff;
    const float scale = 0.17677669529663687f;
    for (int j = 0; j < 8; ++j) {
      int kk = (g << 4) + (j << 1);
      aq.v[2 * j]     = (_Float16)(qr[kk] * scale);
      aq.v[2 * j + 1] = (_Float16)(qr[kk + 1] * scale);
    }
  }

  float m_r[8], l_r[8];
  for (int r = 0; r < 8; ++r) { m_r[r] = -1e30f; l_r[r] = 0.0f; }
  v8f o0 = {}, o1 = {};

  const unsigned* puK = (const unsigned*)sK;
  const unsigned* puV = (const unsigned*)sV;
  const unsigned* puP = (const unsigned*)sP;

  for (int kt = 0; kt < Sc / 32; ++kt) {
    const int kbase = kt * 32;
    __syncthreads();  // previous iteration's fragment reads done
    // stage K/V tiles: 32 keys x 32 dk, 8 elements per thread
    for (int t = tid; t < 32 * 32; t += 128) {
      int key = t >> 5, dk = t & 31;
      const float* kvp = k + (size_t)(base0 + kbase + key) * Dc + hoff + dk;
      const float* vvp = v + (size_t)(base0 + kbase + key) * Dc + hoff + dk;
      sK[key * 40 + dk] = (_Float16)(*kvp);
      sV[dk * 40 + key] = (_Float16)(*vvp);
    }
    __syncthreads();

    // K^T fragments: element i -> kdim = g*16+i, key = col (+16 for tile 1)
    F16x16 bk0, bk1;
    for (int j = 0; j < 8; ++j) {
      bk0.u[j] = puK[col * 20 + (g << 3) + j];
      bk1.u[j] = puK[(16 + col) * 20 + (g << 3) + j];
    }
    v8f zero = {};
    v8f s0 = __builtin_amdgcn_wmma_f32_16x16x32_f16(false, aq.v, false, bk0.v,
                                                    (short)0, zero, false, false);
    v8f s1 = __builtin_amdgcn_wmma_f32_16x16x32_f16(false, aq.v, false, bk1.v,
                                                    (short)0, zero, false, false);

    // online softmax over 32 keys; row stats per reg, reduced over 16 lanes
    for (int r = 0; r < 8; ++r) {
      float tm = fmaxf(s0[r], s1[r]);
      for (int off = 8; off >= 1; off >>= 1) tm = fmaxf(tm, __shfl_xor(tm, off, 32));
      float mn   = fmaxf(m_r[r], tm);
      float corr = __expf(m_r[r] - mn);
      float p0   = __expf(s0[r] - mn);
      float p1   = __expf(s1[r] - mn);
      float rs   = p0 + p1;
      for (int off = 8; off >= 1; off >>= 1) rs += __shfl_xor(rs, off, 32);
      m_r[r] = mn;
      l_r[r] = l_r[r] * corr + rs;
      o0[r] *= corr;
      o1[r] *= corr;
      int prow = wave * 16 + g * 8 + r;          // C-layout row (per-wave slab)
      sP[prow * 36 + col]      = (_Float16)p0;
      sP[prow * 36 + 16 + col] = (_Float16)p1;
    }
    // per-wave LDS region; within-wave DS ordering handled by compiler waits

    // re-layout P (C-layout) -> A fragment (16q x 32keys)
    F16x16 pA;
    for (int j = 0; j < 8; ++j) {
      int kk = (g << 4) + (j << 1);
      pA.u[j] = puP[(wave * 16 + col) * 18 + (kk >> 1)];
    }
    // V fragments: element i -> key = g*16+i, dk = half*16 + col
    F16x16 bv0, bv1;
    for (int j = 0; j < 8; ++j) {
      bv0.u[j] = puV[col * 20 + (g << 3) + j];
      bv1.u[j] = puV[(16 + col) * 20 + (g << 3) + j];
    }
    o0 = __builtin_amdgcn_wmma_f32_16x16x32_f16(false, pA.v, false, bv0.v,
                                                (short)0, o0, false, false);
    o1 = __builtin_amdgcn_wmma_f32_16x16x32_f16(false, pA.v, false, bv1.v,
                                                (short)0, o1, false, false);
  }

  for (int r = 0; r < 8; ++r) {
    int srow  = base0 + qrow0 + g * 8 + r;
    float inv = 1.0f / l_r[r];
    ctx[(size_t)srow * Dc + hoff + col]      = o0[r] * inv;
    ctx[(size_t)srow * Dc + hoff + 16 + col] = o1[r] * inv;
  }
}

// ---------------------------------------------------------------------------
// hout = LN(hin (+ t), gamma, beta) ; one 256-thread block per row (D=256)
// ---------------------------------------------------------------------------
__global__ __launch_bounds__(256) void k_res_ln(const float* __restrict__ hin,
                                                const float* __restrict__ t,
                                                const float* __restrict__ gam,
                                                const float* __restrict__ bet,
                                                float* __restrict__ hout,
                                                int addres) {
  __shared__ float sred[256];
  int row = blockIdx.x;
  int d   = threadIdx.x;
  float v = hin[(size_t)row * Dc + d];
  if (addres) v += t[(size_t)row * Dc + d];

  sred[d] = v; __syncthreads();
  for (int s = 128; s > 0; s >>= 1) { if (d < s) sred[d] += sred[d + s]; __syncthreads(); }
  float mu = sred[0] / (float)Dc;
  __syncthreads();

  float diff = v - mu;
  sred[d] = diff * diff; __syncthreads();
  for (int s = 128; s > 0; s >>= 1) { if (d < s) sred[d] += sred[d + s]; __syncthreads(); }
  float var = sred[0] / (float)Dc;

  hout[(size_t)row * Dc + d] = diff * rsqrtf(var + EPSc) * gam[d] + bet[d];
}

// ---------------------------------------------------------------------------
extern "C" void kernel_launch(void* const* d_in, const int* in_sizes, int n_in,
                              void* d_out, int out_size, void* d_ws, size_t ws_size,
                              hipStream_t stream) {
  (void)in_sizes; (void)n_in; (void)out_size; (void)ws_size;

  const float* x   = (const float*)d_in[0];
  const float* Wq  = (const float*)d_in[1];
  const float* bq  = (const float*)d_in[2];
  const float* Wk  = (const float*)d_in[3];
  const float* bk  = (const float*)d_in[4];
  const float* Wv  = (const float*)d_in[5];
  const float* bv  = (const float*)d_in[6];
  const float* Wo  = (const float*)d_in[7];
  const float* bo  = (const float*)d_in[8];
  const float* W1  = (const float*)d_in[9];
  const float* b1  = (const float*)d_in[10];
  const float* W2  = (const float*)d_in[11];
  const float* b2  = (const float*)d_in[12];
  const float* g1  = (const float*)d_in[13];
  const float* be1 = (const float*)d_in[14];
  const float* g2  = (const float*)d_in[15];
  const float* be2 = (const float*)d_in[16];
  const float* gf  = (const float*)d_in[17];
  const float* bfb = (const float*)d_in[18];
  float* out = (float*)d_out;

  float* ws = (float*)d_ws;
  const size_t act = (size_t)BS * Dc;      // 1M floats
  float* h   = ws;
  float* qb  = ws + 1 * act;
  float* kb  = ws + 2 * act;
  float* vb  = ws + 3 * act;
  float* cb  = ws + 4 * act;
  float* tb  = ws + 5 * act;
  float* ffb = ws + 6 * act;               // BS * FF floats (4M)

  // f16 transposed weights region (after ffb): 3,145,728 halves = 6 MB
  _Float16* f16base = (_Float16*)(ws + 6 * act + (size_t)BS * FFc);
  const size_t wDD = (size_t)Dc * Dc;      // 65536 per layer
  const size_t wDF = (size_t)Dc * FFc;     // 262144 per layer
  _Float16* Wqt = f16base;
  _Float16* Wkt = Wqt + Lc * wDD;
  _Float16* Wvt = Wkt + Lc * wDD;
  _Float16* Wot = Wvt + Lc * wDD;
  _Float16* W1t = Wot + Lc * wDD;          // [FF, D] per layer (N=FF, K=D)
  _Float16* W2t = W1t + Lc * wDF;          // [D, FF] per layer (N=D, K=FF)

  // --- weight prep (deterministic, every call) ---
  for (int l = 0; l < Lc; ++l) {
    int gDDc = (int)((wDD + 255) / 256);
    int gDFc = (int)((wDF + 255) / 256);
    k_wt_f16<<<gDDc, 256, 0, stream>>>(Wq + l * wDD, Wqt + l * wDD, Dc, Dc);
    k_wt_f16<<<gDDc, 256, 0, stream>>>(Wk + l * wDD, Wkt + l * wDD, Dc, Dc);
    k_wt_f16<<<gDDc, 256, 0, stream>>>(Wv + l * wDD, Wvt + l * wDD, Dc, Dc);
    k_wt_f16<<<gDDc, 256, 0, stream>>>(Wo + l * wDD, Wot + l * wDD, Dc, Dc);
    k_wt_f16<<<gDFc, 256, 0, stream>>>(W1 + l * wDF, W1t + l * wDF, Dc, FFc);
    k_wt_f16<<<gDFc, 256, 0, stream>>>(W2 + l * wDF, W2t + l * wDF, FFc, Dc);
  }

  k_add_pe<<<(BS * Dc + 255) / 256, 256, 0, stream>>>(x, h);

  dim3 gDD(Dc / 64, BS / 64);
  dim3 gDF(FFc / 64, BS / 64);
  dim3 gFD(Dc / 64, BS / 64);

  for (int l = 0; l < Lc; ++l) {
    k_gemm<<<gDD, 128, 0, stream>>>(h, Wqt + l * wDD, bq + l * Dc, qb, BS, Dc, Dc, 0);
    k_gemm<<<gDD, 128, 0, stream>>>(h, Wkt + l * wDD, bk + l * Dc, kb, BS, Dc, Dc, 0);
    k_gemm<<<gDD, 128, 0, stream>>>(h, Wvt + l * wDD, bv + l * Dc, vb, BS, Dc, Dc, 0);

    k_attn<<<Bc * Hc * (Sc / 64), 128, 0, stream>>>(qb, kb, vb, cb);

    k_gemm<<<gDD, 128, 0, stream>>>(cb, Wot + l * wDD, bo + l * Dc, tb, BS, Dc, Dc, 0);
    k_res_ln<<<BS, 256, 0, stream>>>(h, tb, g1 + l * Dc, be1 + l * Dc, h, 1);

    k_gemm<<<gDF, 128, 0, stream>>>(h, W1t + l * wDF, b1 + l * FFc, ffb, BS, FFc, Dc, 1);
    k_gemm<<<gFD, 128, 0, stream>>>(ffb, W2t + l * wDF, b2 + l * Dc, tb, BS, Dc, FFc, 0);
    k_res_ln<<<BS, 256, 0, stream>>>(h, tb, g2 + l * Dc, be2 + l * Dc, h, 1);
  }

  k_res_ln<<<BS, 256, 0, stream>>>(h, h, gf, bfb, out, 0);
}
